// OBBAlignConv_70609262346550
// MI455X (gfx1250) — matrix operand order repaired and checked by reading, not compiled, not run on hardware
//
#include <hip/hip_runtime.h>
#include <stdint.h>

#define CIN    256
#define COUT   256
#define HH     128
#define WW     128
#define HWSZ   16384
#define NBATCH 2
#define KDIM   2304
#define BM     128
#define BN     128
#define BK     32
#define NSTEP  72          // KDIM / BK

typedef __attribute__((ext_vector_type(16))) __bf16 v16bf;
typedef __attribute__((ext_vector_type(8)))  float  v8f;

__device__ __forceinline__ uint32_t f2bf_u(float f) {
  uint32_t u = __float_as_uint(f);
  u += 0x7FFFu + ((u >> 16) & 1u);   // round-to-nearest-even
  return u >> 16;
}
__device__ __forceinline__ float bf2f(uint32_t h) {
  return __uint_as_float(h << 16);
}

// ---------------------------------------------------------------------------
// Kernel 1: x [N,Cin,H,W] f32  ->  x_nhwc [N,H*W,Cin] bf16  (LDS transpose)
// ---------------------------------------------------------------------------
__global__ __launch_bounds__(256)
void pack_x_kernel(const float* __restrict__ x, uint16_t* __restrict__ xn) {
  __shared__ uint16_t tile[32 * 132];
  const int n  = blockIdx.z;
  const int c0 = blockIdx.y * 32;
  const int p0 = blockIdx.x * 128;
  const int t  = threadIdx.x;
#pragma unroll
  for (int i = 0; i < 16; ++i) {
    int flat = t + i * 256;          // 0..4095
    int c = flat >> 7;               // 0..31
    int p = flat & 127;              // coalesced along hw
    float v = x[((size_t)(n * CIN + c0 + c) << 14) + p0 + p];
    tile[c * 132 + p] = (uint16_t)f2bf_u(v);
  }
  __syncthreads();
#pragma unroll
  for (int i = 0; i < 16; ++i) {
    int flat = t + i * 256;
    int c = flat & 31;               // coalesced along channels
    int p = flat >> 5;
    xn[((size_t)(n * HWSZ + p0 + p) << 8) + c0 + c] = tile[c * 132 + p];
  }
}

// ---------------------------------------------------------------------------
// Kernel 2: weight -> bf16 packed in B-fragment order:
//   word index = ((kstep*16 + nfrag)*32 + lane)*8 + v
//   col = nfrag*16 + (lane&15); K pair base k0 = 2*v + 16*(lane>>4)
// ---------------------------------------------------------------------------
__global__ __launch_bounds__(256)
void pack_w_kernel(const float* __restrict__ w, uint32_t* __restrict__ bf) {
  int idx = blockIdx.x * 256 + threadIdx.x;   // < 294912
  int v   = idx & 7;
  int L   = (idx >> 3) & 31;
  int nf  = (idx >> 8) & 15;
  int ks  = idx >> 12;                        // 0..71
  int k0  = v * 2 + (L >> 4) * 16;
  int kg  = ks * 32 + k0;
  int col = nf * 16 + (L & 15);
  int kk  = kg >> 8;
  int c   = kg & 255;
  size_t base = ((size_t)(col * CIN + c)) * 9 + kk;
  uint32_t lo = f2bf_u(w[base]);
  uint32_t hi = f2bf_u(w[base + 9]);          // c+1, same kk
  bf[idx] = lo | (hi << 16);
}

// ---------------------------------------------------------------------------
// Kernel 3: obb -> sampling anchors pos[n][kk][hw] = (y_anchor, x_anchor)
// ---------------------------------------------------------------------------
__global__ __launch_bounds__(256)
void pos_kernel(const float* __restrict__ obb, const int* __restrict__ stride_p,
                float2* __restrict__ pos) {
  int t  = blockIdx.x * 256 + threadIdx.x;    // < 32768
  int n  = t >> 14;
  int hw = t & (HWSZ - 1);
  float s  = (float)(*stride_p);
  float rs = 1.0f / s;
  const float* ob = obb + (size_t)n * 5 * HWSZ + hw;
  float xc = ob[0 * HWSZ] * rs, yc = ob[1 * HWSZ] * rs;
  float wb = ob[2 * HWSZ] * rs, hb = ob[3 * HWSZ] * rs;
  float a  = ob[4 * HWSZ];
  float cs = cosf(a), sn = sinf(a);
  float dw = wb * (1.0f / 3.0f), dh = hb * (1.0f / 3.0f);
#pragma unroll
  for (int kk = 0; kk < 9; ++kk) {
    float fx = (float)(kk % 3 - 1);
    float fy = (float)(kk / 3 - 1);
    float pxr = dw * fx, pyr = dh * fy;
    float xa = cs * pxr - sn * pyr + xc;
    float ya = sn * pxr + cs * pyr + yc;
    pos[((size_t)(n * 9 + kk) << 14) + hw] = make_float2(ya, xa);
  }
}

// ---------------------------------------------------------------------------
// Kernel 4: implicit GEMM, bf16 WMMA, double-buffered A tile, pipelined
// corner prefetch, pointer-increment B addressing.
// ---------------------------------------------------------------------------
__global__ __launch_bounds__(256)
void gemm_kernel(const uint16_t* __restrict__ xn,
                 const uint32_t* __restrict__ bfrag,
                 const float2*  __restrict__ posb,
                 float*         __restrict__ out) {
  // two A buffers: 128 rows x 20 dwords (80B pitch, conflict-free b128)
  __shared__ __align__(16) uint32_t At32[2 * BM * 20];

  const int tid   = threadIdx.x;
  const int lane  = tid & 31;
  const int wv    = tid >> 5;
  const int wm    = wv & 3;        // rows wm*32
  const int wn    = wv >> 2;       // cols wn*64
  const int m0    = blockIdx.x * BM;
  const int halfL = lane >> 4;
  const int l15   = lane & 15;

  // ---- staging identity: 2 threads per row ----
  const int r  = tid >> 1;
  const int hf = tid & 1;
  const int mrow = m0 + r;
  const int n_r  = mrow >> 14;
  const int hw_r = mrow & (HWSZ - 1);
  const float2* posp = posb + ((size_t)n_r * 9 << 14) + hw_r;
  const uint16_t* xnb = xn + ((size_t)n_r << 22) + hf * 16;   // n*HW*CIN, + half

  // ---- B pointer: byte addr = 16384*ks + 1024*nf + 32*lane ----
  const char* bptr = (const char*)bfrag +
      1024 * (blockIdx.y * 8 + wn * 4) + 32 * lane;

  v8f acc[2][4];
#pragma unroll
  for (int f = 0; f < 2; ++f)
#pragma unroll
    for (int g = 0; g < 4; ++g)
#pragma unroll
      for (int j = 0; j < 8; ++j) acc[f][g][j] = 0.0f;

  // per-row corner pointers (advance 64B per step) + bilinear weights
  const uint4 *q0, *q1, *q2, *q3;
  float wt0, wt1, wt2, wt3;

  auto setup_row = [&](int kk) {
    float2 p = posp[(size_t)kk << 14];
    float py = p.x, px = p.y;
    float y0f = floorf(py), x0f = floorf(px);
    int y0 = (int)y0f, x0 = (int)x0f;
    float ly = py - y0f, lx = px - x0f;
    float wy0 = 1.0f - ly, wx0 = 1.0f - lx;
    float wgt[4] = { wy0 * wx0, wy0 * lx, ly * wx0, ly * lx };
    int iy[4] = { y0, y0, y0 + 1, y0 + 1 };
    int ix[4] = { x0, x0 + 1, x0, x0 + 1 };
    const uint4* qq[4];
    float ww[4];
#pragma unroll
    for (int s2 = 0; s2 < 4; ++s2) {
      bool valid = (iy[s2] >= 0) && (iy[s2] < HH) && (ix[s2] >= 0) && (ix[s2] < WW);
      int yc = min(max(iy[s2], 0), HH - 1);
      int xc = min(max(ix[s2], 0), WW - 1);
      qq[s2] = (const uint4*)(xnb + ((yc * WW + xc) << 8));
      ww[s2] = valid ? wgt[s2] : 0.0f;
    }
    q0 = qq[0]; q1 = qq[1]; q2 = qq[2]; q3 = qq[3];
    wt0 = ww[0]; wt1 = ww[1]; wt2 = ww[2]; wt3 = ww[3];
  };

  // ---- prologue: prefetch step 0 ----
  setup_row(0);
  uint4 pa0 = q0[0], pa1 = q0[1];
  uint4 pb0 = q1[0], pb1 = q1[1];
  uint4 pc0 = q2[0], pc1 = q2[1];
  uint4 pd0 = q3[0], pd1 = q3[1];

  for (int s = 0; s < NSTEP; ++s) {
    const int bufo = (s & 1) * (BM * 20);

    // ---- convert prefetched corners -> bf16 pairs -> LDS ----
    {
      uint32_t ca[8] = { pa0.x, pa0.y, pa0.z, pa0.w, pa1.x, pa1.y, pa1.z, pa1.w };
      uint32_t cb[8] = { pb0.x, pb0.y, pb0.z, pb0.w, pb1.x, pb1.y, pb1.z, pb1.w };
      uint32_t cc[8] = { pc0.x, pc0.y, pc0.z, pc0.w, pc1.x, pc1.y, pc1.z, pc1.w };
      uint32_t cd[8] = { pd0.x, pd0.y, pd0.z, pd0.w, pd1.x, pd1.y, pd1.z, pd1.w };
      uint32_t o[8];
#pragma unroll
      for (int j = 0; j < 8; ++j) {
        float lo = wt0 * bf2f(ca[j] & 0xffffu) + wt1 * bf2f(cb[j] & 0xffffu)
                 + wt2 * bf2f(cc[j] & 0xffffu) + wt3 * bf2f(cd[j] & 0xffffu);
        float hi = wt0 * bf2f(ca[j] >> 16) + wt1 * bf2f(cb[j] >> 16)
                 + wt2 * bf2f(cc[j] >> 16) + wt3 * bf2f(cd[j] >> 16);
        o[j] = f2bf_u(lo) | (f2bf_u(hi) << 16);
      }
      uint4* dst = (uint4*)(At32 + bufo + r * 20 + hf * 8);
      dst[0] = make_uint4(o[0], o[1], o[2], o[3]);
      dst[1] = make_uint4(o[4], o[5], o[6], o[7]);
    }

    // ---- issue prefetch for step s+1 (overlaps barrier + WMMA below) ----
    if (s < NSTEP - 1) {
      if (((s + 1) & 7) == 0) {
        setup_row((s + 1) >> 3);        // new kk: fresh pointers + weights
      } else {
        q0 += 4; q1 += 4; q2 += 4; q3 += 4;   // +64B: next 32 channels
      }
      pa0 = q0[0]; pa1 = q0[1];
      pb0 = q1[0]; pb1 = q1[1];
      pc0 = q2[0]; pc1 = q2[1];
      pd0 = q3[0]; pd1 = q3[1];
    }

    __syncthreads();   // single barrier/step (safe with double buffer)

    // ---- fragment loads + 8 WMMA ----
    union Frag { uint4 q[2]; v16bf v; };
    Frag Af[2], Bf[4];
#pragma unroll
    for (int f = 0; f < 2; ++f) {
      int row = wm * 32 + f * 16 + l15;
      const uint4* ap = (const uint4*)(At32 + bufo + row * 20 + halfL * 8);
      Af[f].q[0] = ap[0];   // v0..3
      Af[f].q[1] = ap[1];   // v4..7
    }
#pragma unroll
    for (int g = 0; g < 4; ++g) {
      Bf[g].q[0] = *(const uint4*)(bptr + g * 1024);
      Bf[g].q[1] = *(const uint4*)(bptr + g * 1024 + 16);
    }
    bptr += 16384;
#pragma unroll
    for (int f = 0; f < 2; ++f)
#pragma unroll
      for (int g = 0; g < 4; ++g)
        acc[f][g] = __builtin_amdgcn_wmma_f32_16x16x32_bf16(
            false, Af[f].v, false, Bf[g].v, (short)0, acc[f][g], false, false);
  }

  // ---- epilogue: C layout VGPR j -> M = j + 8*half, N = lane&15 ----
#pragma unroll
  for (int f = 0; f < 2; ++f) {
    int mbase = m0 + wm * 32 + f * 16 + halfL * 8;
#pragma unroll
    for (int g = 0; g < 4; ++g) {
      int cout = blockIdx.y * BN + wn * 64 + g * 16 + l15;
#pragma unroll
      for (int j = 0; j < 8; ++j) {
        int m  = mbase + j;
        int n  = m >> 14;
        int hw = m & (HWSZ - 1);
        out[((size_t)(n * COUT + cout) << 14) + hw] = acc[f][g][j];
      }
    }
  }
}

// ---------------------------------------------------------------------------
extern "C" void kernel_launch(void* const* d_in, const int* in_sizes, int n_in,
                              void* d_out, int out_size, void* d_ws, size_t ws_size,
                              hipStream_t stream) {
  (void)in_sizes; (void)n_in; (void)out_size; (void)ws_size;
  const float* x   = (const float*)d_in[0];   // [2,256,128,128]
  const float* obb = (const float*)d_in[1];   // [2,5,128,128]
  const float* wgt = (const float*)d_in[2];   // [256,256,3,3]
  const int*   str = (const int*)d_in[3];     // scalar stride
  float* out = (float*)d_out;                 // [2,256,128,128]

  char* ws = (char*)d_ws;
  uint16_t* xn   = (uint16_t*)(ws);                        // 16.78 MB
  uint32_t* bfr  = (uint32_t*)(ws + 16777216);             //  1.18 MB
  float2*   posb = (float2*)  (ws + 16777216 + 1179648);   //  2.36 MB

  pack_x_kernel<<<dim3(HWSZ / 128, CIN / 32, NBATCH), 256, 0, stream>>>(x, xn);
  pack_w_kernel<<<dim3(294912 / 256), 256, 0, stream>>>(wgt, bfr);
  pos_kernel<<<dim3((NBATCH * HWSZ) / 256), 256, 0, stream>>>(obb, str, posb);
  gemm_kernel<<<dim3((NBATCH * HWSZ) / BM, COUT / BN), 256, 0, stream>>>(xn, bfr, posb, out);
}
